// Transformer_89378269430065
// MI455X (gfx1250) — matrix operand order repaired and checked by reading, not compiled, not run on hardware
//
#include <hip/hip_runtime.h>

// ---------------------------------------------------------------------------
// GPT forward (B=4,T=1024,E=1024,H=16,HS=64,L=8,V=32000) for MI455X (gfx1250).
// All GEMMs: v_wmma_f32_16x16x32_f16 (wave32), fp32 accumulate.
// Double-buffered LDS tiles (one barrier per K-step, global fetch overlapped
// with WMMA), b128 fragment loads, transposed-B staging, padded LDS rows.
// ---------------------------------------------------------------------------

typedef __attribute__((ext_vector_type(16))) _Float16 v16h;
typedef __attribute__((ext_vector_type(8)))  _Float16 v8h;
typedef __attribute__((ext_vector_type(4)))  _Float16 v4h;
typedef __attribute__((ext_vector_type(8)))  float    v8f;
typedef __attribute__((ext_vector_type(4)))  float    v4f;

constexpr int Vv = 32000, Ee = 1024, Hh = 16, HSs = 64, Ll = 8, Tt = 1024, Bb = 4;
constexpr int BT = Bb * Tt;   // 4096 activation rows

// ---------------------------------------------------------------- WMMA core
__device__ __forceinline__ v8f wmma16(v16h a, v16h b, v8f c) {
  return __builtin_amdgcn_wmma_f32_16x16x32_f16(
      false, a, false, b, (short)0, c, false, false);
}

// A-fragment (16x32 f16): two contiguous 8-half (16B) runs per lane.
__device__ __forceinline__ v16h a_frag(const _Float16* base, int ld,
                                       int row0, int k0, int lane) {
  const _Float16* p = base + (row0 + (lane & 15)) * ld + k0 + ((lane >> 4) << 3);
  v8h lo = *(const v8h*)p;
  v8h hi = *(const v8h*)(p + 16);
  v16h a;
#pragma unroll
  for (int i = 0; i < 8; ++i) { a[i] = lo[i]; a[8 + i] = hi[i]; }
  return a;
}

// Same A-fragment from fp32 LDS (score strip), converted on the fly.
__device__ __forceinline__ v16h a_frag_f32(const float* base, int ld,
                                           int row0, int k0, int lane) {
  const float* p = base + (size_t)(row0 + (lane & 15)) * ld + k0 + ((lane >> 4) << 3);
  v4f x0 = *(const v4f*)p;
  v4f x1 = *(const v4f*)(p + 4);
  v4f x2 = *(const v4f*)(p + 16);
  v4f x3 = *(const v4f*)(p + 20);
  v16h a;
#pragma unroll
  for (int i = 0; i < 4; ++i) {
    a[i]      = (_Float16)x0[i];
    a[4 + i]  = (_Float16)x1[i];
    a[8 + i]  = (_Float16)x2[i];
    a[12 + i] = (_Float16)x3[i];
  }
  return a;
}

// B-fragment (32x16 f16) from a TRANSPOSED tile baseT[n][k]:
// two 8-half runs per lane at [col][k0+8*hi..+7] and [col][k0+16+8*hi..+7].
__device__ __forceinline__ v16h b_fragT(const _Float16* baseT, int ld,
                                        int col0, int k0, int lane) {
  const _Float16* p = baseT + (col0 + (lane & 15)) * ld + k0 + ((lane >> 4) << 3);
  v8h lo = *(const v8h*)p;
  v8h hi = *(const v8h*)(p + 16);
  v16h b;
#pragma unroll
  for (int i = 0; i < 8; ++i) { b[i] = lo[i]; b[8 + i] = hi[i]; }
  return b;
}

// ---------------------------------------------------------------- embedding
__global__ void __launch_bounds__(256)
embed_kernel(const int* __restrict__ inputs, const float* __restrict__ tok,
             const float* __restrict__ pos, float* __restrict__ x) {
  int row = blockIdx.x;
  int t = row % Tt;
  int id = inputs[row];
  int i = threadIdx.x * 4;         // Ee == 256*4
  v4f tv = *(const v4f*)&tok[(size_t)id * Ee + i];
  v4f pv = *(const v4f*)&pos[(size_t)t * Ee + i];
  *(v4f*)&x[(size_t)row * Ee + i] = tv + pv;
}

// ---------------------------------------------------------------- layernorm
__global__ void __launch_bounds__(256)
layernorm_kernel(const float* __restrict__ x, const float* __restrict__ g,
                 const float* __restrict__ b, float* __restrict__ y) {
  __shared__ float red[256];
  int row = blockIdx.x, tid = threadIdx.x;
  int i = tid * 4;
  v4f xv = *(const v4f*)&x[(size_t)row * Ee + i];

  float s = xv[0] + xv[1] + xv[2] + xv[3];
  red[tid] = s; __syncthreads();
  for (int off = 128; off > 0; off >>= 1) {
    if (tid < off) red[tid] += red[tid + off];
    __syncthreads();
  }
  float mean = red[0] * (1.0f / Ee);
  __syncthreads();

  v4f d; float v = 0.f;
#pragma unroll
  for (int j = 0; j < 4; ++j) { d[j] = xv[j] - mean; v += d[j] * d[j]; }
  red[tid] = v; __syncthreads();
  for (int off = 128; off > 0; off >>= 1) {
    if (tid < off) red[tid] += red[tid + off];
    __syncthreads();
  }
  float inv = rsqrtf(red[0] * (1.0f / Ee) + 1e-5f);

  v4f gv = *(const v4f*)&g[i];
  v4f bv = *(const v4f*)&b[i];
  v4f yv;
#pragma unroll
  for (int j = 0; j < 4; ++j) yv[j] = d[j] * inv * gv[j] + bv[j];
  *(v4f*)&y[(size_t)row * Ee + i] = yv;
}

// ------------------------------------------------------------- generic GEMM
// C[M,N] = act(A[M,K]*B[K,N] + bias) + resid.  M%128==0, N%64==0, K%64==0.
// bHeads=1: B is [H][K][64] (QKV weights, n = h*64+s).  act=1: ReLU.
// Double-buffered: stage->barrier->(fetch next)->compute; 1 barrier / K-step.
#define BM 128
#define BN 64
#define BK 64
#define LDA 72
#define LDB 72
__global__ void __launch_bounds__(256)
gemm_kernel(const float* __restrict__ A, const float* __restrict__ B,
            const float* __restrict__ bias, const float* __restrict__ resid,
            float* __restrict__ C, int M, int N, int K, int bHeads, int act) {
  __shared__ __align__(16) _Float16 As[2][BM * LDA];    // 2 x 18 KB
  __shared__ __align__(16) _Float16 BsT[2][BN * LDB];   // 2 x  9 KB
  int tid = threadIdx.x, lane = tid & 31, wave = tid >> 5;
  int m0 = blockIdx.y * BM, n0 = blockIdx.x * BN;

  v8f acc[4] = {{}, {}, {}, {}};

  int bn = tid & 63, bkg = tid >> 6;
  int gn = n0 + bn;
  size_t bColBase = bHeads ? ((size_t)(gn >> 6) * K * 64 + (gn & 63))
                           : (size_t)gn;
  size_t bStride = bHeads ? 64 : (size_t)N;

  v4f ta[8];
  float tb[16];
  // prologue fetch (k0 = 0)
#pragma unroll
  for (int i = 0; i < 8; ++i) {
    int e = (i * 256 + tid) * 4, r = e >> 6, c = e & 63;
    ta[i] = *(const v4f*)&A[(size_t)(m0 + r) * K + c];
  }
#pragma unroll
  for (int i = 0; i < 16; ++i)
    tb[i] = B[bColBase + (size_t)(bkg * 16 + i) * bStride];

  int buf = 0;
  for (int k0 = 0; k0 < K; k0 += BK) {
    // ---- stage registers -> LDS[buf]
    _Float16* Ab = As[buf];
#pragma unroll
    for (int i = 0; i < 8; ++i) {
      int e = (i * 256 + tid) * 4, r = e >> 6, c = e & 63;
      v4h h;
#pragma unroll
      for (int j = 0; j < 4; ++j) h[j] = (_Float16)ta[i][j];
      *(v4h*)&Ab[r * LDA + c] = h;
    }
    {
      v8h p0, p1;
#pragma unroll
      for (int i = 0; i < 8; ++i) { p0[i] = (_Float16)tb[i]; p1[i] = (_Float16)tb[8 + i]; }
      _Float16* dst = &BsT[buf][bn * LDB + bkg * 16];
      *(v8h*)dst = p0;
      *(v8h*)(dst + 8) = p1;
    }
    __syncthreads();
    // ---- issue next tile's global loads (hidden behind WMMAs below)
    if (k0 + BK < K) {
#pragma unroll
      for (int i = 0; i < 8; ++i) {
        int e = (i * 256 + tid) * 4, r = e >> 6, c = e & 63;
        ta[i] = *(const v4f*)&A[(size_t)(m0 + r) * K + (k0 + BK) + c];
      }
#pragma unroll
      for (int i = 0; i < 16; ++i)
        tb[i] = B[bColBase + (size_t)(k0 + BK + bkg * 16 + i) * bStride];
    }
    // ---- compute: wave = 16-row strip x 4 col tiles, A-frag reused 4x
#pragma unroll
    for (int kk = 0; kk < BK; kk += 32) {
      v16h a = a_frag(Ab, LDA, wave * 16, kk, lane);
#pragma unroll
      for (int tn = 0; tn < 4; ++tn) {
        v16h b = b_fragT(BsT[buf], LDB, tn * 16, kk, lane);
        acc[tn] = wmma16(a, b, acc[tn]);
      }
    }
    buf ^= 1;
  }

  int rbase = m0 + wave * 16 + ((lane >> 4) << 3);
#pragma unroll
  for (int tn = 0; tn < 4; ++tn) {
    int col = n0 + tn * 16 + (lane & 15);
    float bs = bias ? bias[col] : 0.f;
#pragma unroll
    for (int r = 0; r < 8; ++r) {
      float vv = acc[tn][r] + bs;
      if (act == 1) vv = fmaxf(vv, 0.f);
      size_t idx = (size_t)(rbase + r) * N + col;
      if (resid) vv += resid[idx];
      C[idx] = vv;
    }
  }
}

// ---------------------------------------------------------------- attention
// One block per (b, h, 32 query rows). q/k/v/o layout: [B][T][H][HS].
// LDS: S 32x(T+4) f32 | Q 32x72 f16 | KV tile 2x 64x72 f16 (double buffer).
#define LDS_S (Tt + 4)
#define LDH 72
__global__ void __launch_bounds__(256)
attn_kernel(const float* __restrict__ q, const float* __restrict__ k,
            const float* __restrict__ v, float* __restrict__ o, float scale) {
  extern __shared__ __align__(16) char smem[];
  float*    S  = (float*)smem;                    // [32][LDS_S]
  _Float16* Qs = (_Float16*)(S + 32 * LDS_S);     // [32][LDH]
  _Float16* KV = Qs + 32 * LDH;                   // [2][64][LDH]

  int bh = blockIdx.x, b = bh / Hh, h = bh % Hh;
  int q0 = blockIdx.y * 32;
  int tid = threadIdx.x, lane = tid & 31, wave = tid >> 5;
  int tm = wave >> 2, tn = wave & 3;              // 2x4 tiles over 32x64

  // stage Q strip (32x64)
#pragma unroll
  for (int i = 0; i < 2; ++i) {
    int e = (i * 256 + tid) * 4, r = e >> 6, c = e & 63;
    v4f t = *(const v4f*)&q[(((size_t)b * Tt + q0 + r) * Hh + h) * HSs + c];
    v4h hh;
#pragma unroll
    for (int j = 0; j < 4; ++j) hh[j] = (_Float16)t[j];
    *(v4h*)&Qs[r * LDH + c] = hh;
  }

  // ---- phase 1: S = scale * Q K^T  (double-buffered K tiles)
  v4f tk[4];
#pragma unroll
  for (int i = 0; i < 4; ++i) {
    int e = (i * 256 + tid) * 4, r = e >> 6, c = e & 63;
    tk[i] = *(const v4f*)&k[(((size_t)b * Tt + r) * Hh + h) * HSs + c];
  }
  int buf = 0;
  for (int jt = 0; jt < Tt / 64; ++jt) {
    _Float16* KVb = KV + buf * (64 * LDH);
#pragma unroll
    for (int i = 0; i < 4; ++i) {
      int e = (i * 256 + tid) * 4, r = e >> 6, c = e & 63;
      v4h hh;
#pragma unroll
      for (int j = 0; j < 4; ++j) hh[j] = (_Float16)tk[i][j];
      *(v4h*)&KVb[r * LDH + c] = hh;
    }
    __syncthreads();   // also covers Qs on first iteration
    if (jt + 1 < Tt / 64) {
      int j0n = (jt + 1) * 64;
#pragma unroll
      for (int i = 0; i < 4; ++i) {
        int e = (i * 256 + tid) * 4, r = e >> 6, c = e & 63;
        tk[i] = *(const v4f*)&k[(((size_t)b * Tt + j0n + r) * Hh + h) * HSs + c];
      }
    }
    v8f c8 = {};
#pragma unroll
    for (int kk = 0; kk < HSs; kk += 32) {
      v16h a = a_frag(Qs, LDH, tm * 16, kk, lane);
      v16h bf = b_fragT(KVb, LDH, tn * 16, kk, lane);  // B = K^T: KV[j][s]
      c8 = wmma16(a, bf, c8);
    }
    int rbase = tm * 16 + ((lane >> 4) << 3);
    int col = jt * 64 + tn * 16 + (lane & 15);
#pragma unroll
    for (int r = 0; r < 8; ++r) S[(rbase + r) * LDS_S + col] = c8[r] * scale;
    buf ^= 1;
  }
  __syncthreads();

  // ---- phase 2: causal softmax (8 threads per row, intra-wave shuffles)
  {
    int row = tid >> 3, sub = tid & 7;
    int qg = q0 + row;
    float* Sr = S + row * LDS_S;
    float mx = -1e30f;
    for (int j = sub; j < Tt; j += 8)
      mx = fmaxf(mx, (j <= qg) ? Sr[j] : -1e30f);
    for (int off = 4; off > 0; off >>= 1) mx = fmaxf(mx, __shfl_xor(mx, off, 8));
    float sum = 0.f;
    for (int j = sub; j < Tt; j += 8) {
      float p = (j <= qg) ? __expf(Sr[j] - mx) : 0.f;
      Sr[j] = p;
      sum += p;
    }
    for (int off = 4; off > 0; off >>= 1) sum += __shfl_xor(sum, off, 8);
    float inv = 1.f / sum;
    for (int j = sub; j < Tt; j += 8) Sr[j] *= inv;
  }
  __syncthreads();

  // ---- phase 3: O = P V  (double-buffered V^T tiles: VsT[s][j])
  v8f acc = {};
  int vs = tid & 63, vjg = tid >> 6;
  float tv[16];
#pragma unroll
  for (int i = 0; i < 16; ++i)
    tv[i] = v[(((size_t)b * Tt + vjg * 16 + i) * Hh + h) * HSs + vs];
  buf = 0;
  for (int jt = 0; jt < Tt / 64; ++jt) {
    _Float16* KVb = KV + buf * (64 * LDH);
    {
      v8h p0, p1;
#pragma unroll
      for (int i = 0; i < 8; ++i) { p0[i] = (_Float16)tv[i]; p1[i] = (_Float16)tv[8 + i]; }
      _Float16* dst = &KVb[vs * LDH + vjg * 16];
      *(v8h*)dst = p0;
      *(v8h*)(dst + 8) = p1;
    }
    __syncthreads();
    if (jt + 1 < Tt / 64) {
      int j0n = (jt + 1) * 64;
#pragma unroll
      for (int i = 0; i < 16; ++i)
        tv[i] = v[(((size_t)b * Tt + j0n + vjg * 16 + i) * Hh + h) * HSs + vs];
    }
#pragma unroll
    for (int kk = 0; kk < 64; kk += 32) {
      v16h a = a_frag_f32(S, LDS_S, tm * 16, jt * 64 + kk, lane);
      v16h bf = b_fragT(KVb, LDH, tn * 16, kk, lane);  // B[j][s] = VsT[s][j]
      acc = wmma16(a, bf, acc);
    }
    buf ^= 1;
  }
  int rbase = tm * 16 + ((lane >> 4) << 3);
  int col = tn * 16 + (lane & 15);
#pragma unroll
  for (int r = 0; r < 8; ++r)
    o[(((size_t)b * Tt + q0 + rbase + r) * Hh + h) * HSs + col] = acc[r];
}

// --------------------------------------------------------------- loss stage
__global__ void __launch_bounds__(256)
rowloss_kernel(const float* __restrict__ logits, const int* __restrict__ targets,
               float* __restrict__ rowloss) {
  constexpr int V4 = (Vv / 1024) * 1024;   // 31744; tail = 256 = blockDim
  __shared__ float red[256];
  int row = blockIdx.x, tid = threadIdx.x;
  const float* lr = logits + (size_t)row * Vv;

  float mx = -1e30f;
  for (int i = tid * 4; i < V4; i += 1024) {
    v4f t = *(const v4f*)&lr[i];
    mx = fmaxf(mx, fmaxf(fmaxf(t[0], t[1]), fmaxf(t[2], t[3])));
  }
  if (V4 + tid < Vv) mx = fmaxf(mx, lr[V4 + tid]);
  red[tid] = mx; __syncthreads();
  for (int off = 128; off > 0; off >>= 1) {
    if (tid < off) red[tid] = fmaxf(red[tid], red[tid + off]);
    __syncthreads();
  }
  mx = red[0]; __syncthreads();

  float s = 0.f;
  for (int i = tid * 4; i < V4; i += 1024) {
    v4f t = *(const v4f*)&lr[i];
    s += __expf(t[0] - mx) + __expf(t[1] - mx) + __expf(t[2] - mx) + __expf(t[3] - mx);
  }
  if (V4 + tid < Vv) s += __expf(lr[V4 + tid] - mx);
  red[tid] = s; __syncthreads();
  for (int off = 128; off > 0; off >>= 1) {
    if (tid < off) red[tid] += red[tid + off];
    __syncthreads();
  }
  if (tid == 0)
    rowloss[row] = -(lr[targets[row]] - mx - __logf(red[0]));
}

__global__ void __launch_bounds__(256)
loss_reduce_kernel(const float* __restrict__ rowloss, float* __restrict__ out) {
  __shared__ float red[256];
  int tid = threadIdx.x;
  float s = 0.f;
  for (int i = tid; i < BT; i += 256) s += rowloss[i];
  red[tid] = s; __syncthreads();
  for (int off = 128; off > 0; off >>= 1) {
    if (tid < off) red[tid] += red[tid + off];
    __syncthreads();
  }
  if (tid == 0) out[0] = red[0] * (1.0f / BT);
}

// ----------------------------------------------------------------- host side
static void launch_gemm(const float* A, const float* B, const float* bias,
                        const float* resid, float* C, int M, int N, int K,
                        int bHeads, int act, hipStream_t s) {
  dim3 grid(N / BN, M / BM);
  gemm_kernel<<<grid, 256, 0, s>>>(A, B, bias, resid, C, M, N, K, bHeads, act);
}

extern "C" void kernel_launch(void* const* d_in, const int* in_sizes, int n_in,
                              void* d_out, int out_size, void* d_ws, size_t ws_size,
                              hipStream_t stream) {
  (void)in_sizes; (void)n_in; (void)out_size; (void)ws_size;

  const int*   inputs  = (const int*)d_in[0];
  const int*   targets = (const int*)d_in[1];
  const float* tok     = (const float*)d_in[2];
  const float* pos     = (const float*)d_in[3];
  const float* Wq      = (const float*)d_in[4];
  const float* Wk      = (const float*)d_in[5];
  const float* Wv      = (const float*)d_in[6];
  const float* Wo      = (const float*)d_in[7];
  const float* bo      = (const float*)d_in[8];
  const float* ln1g    = (const float*)d_in[9];
  const float* ln1b    = (const float*)d_in[10];
  const float* ln2g    = (const float*)d_in[11];
  const float* ln2b    = (const float*)d_in[12];
  const float* W1      = (const float*)d_in[13];
  const float* b1      = (const float*)d_in[14];
  const float* W2      = (const float*)d_in[15];
  const float* b2      = (const float*)d_in[16];
  const float* lnfg    = (const float*)d_in[17];
  const float* lnfb    = (const float*)d_in[18];
  const float* Wlm     = (const float*)d_in[19];
  const float* blm     = (const float*)d_in[20];
  float* out = (float*)d_out;

  float* x   = (float*)d_ws;
  float* hb  = x   + (size_t)BT * Ee;
  float* qb  = hb  + (size_t)BT * Ee;
  float* kb  = qb  + (size_t)BT * Ee;
  float* vb  = kb  + (size_t)BT * Ee;
  float* ob  = vb  + (size_t)BT * Ee;
  float* mid = ob  + (size_t)BT * Ee;
  float* rls = mid + (size_t)BT * 4 * Ee;

  const float scale = 0.03125f;  // E^-0.5 (reference scales by E, not HS)
  const size_t attn_smem = (size_t)32 * LDS_S * sizeof(float)
                         + (size_t)(32 * LDH + 2 * 64 * LDH) * sizeof(_Float16);

  embed_kernel<<<BT, 256, 0, stream>>>(inputs, tok, pos, x);

  for (int l = 0; l < Ll; ++l) {
    layernorm_kernel<<<BT, 256, 0, stream>>>(x, ln1g + (size_t)l * Ee,
                                             ln1b + (size_t)l * Ee, hb);
    const float* Wql = Wq + (size_t)l * Hh * Ee * HSs;
    const float* Wkl = Wk + (size_t)l * Hh * Ee * HSs;
    const float* Wvl = Wv + (size_t)l * Hh * Ee * HSs;
    launch_gemm(hb, Wql, nullptr, nullptr, qb, BT, Ee, Ee, 1, 0, stream);
    launch_gemm(hb, Wkl, nullptr, nullptr, kb, BT, Ee, Ee, 1, 0, stream);
    launch_gemm(hb, Wvl, nullptr, nullptr, vb, BT, Ee, Ee, 1, 0, stream);

    attn_kernel<<<dim3(Bb * Hh, Tt / 32), 256, attn_smem, stream>>>(qb, kb, vb, ob, scale);

    launch_gemm(ob, Wo + (size_t)l * Ee * Ee, bo + (size_t)l * Ee,
                x, x, BT, Ee, Ee, 0, 0, stream);

    layernorm_kernel<<<BT, 256, 0, stream>>>(x, ln2g + (size_t)l * Ee,
                                             ln2b + (size_t)l * Ee, hb);
    launch_gemm(hb, W1 + (size_t)l * Ee * 4 * Ee, b1 + (size_t)l * 4 * Ee,
                nullptr, mid, BT, 4 * Ee, Ee, 0, 1, stream);
    launch_gemm(mid, W2 + (size_t)l * 4 * Ee * Ee, b2 + (size_t)l * Ee,
                x, x, BT, Ee, 4 * Ee, 0, 0, stream);
  }

  layernorm_kernel<<<BT, 256, 0, stream>>>(x, lnfg, lnfb, hb);
  launch_gemm(hb, Wlm, blm, nullptr, out, BT, Vv, Ee, 0, 0, stream);

  rowloss_kernel<<<BT, 256, 0, stream>>>(out, targets, rls);
  loss_reduce_kernel<<<1, 256, 0, stream>>>(rls, out + (size_t)BT * Vv);
}
